// GraphLearningProbSparseAttention_8340826488955
// MI455X (gfx1250) — compile-verified, matrix operand order checked by reading, not run on hardware
//
#include <hip/hip_runtime.h>
#include <hip/hip_bf16.h>
#include <math.h>

typedef __attribute__((ext_vector_type(16))) _Float16 v16h;
typedef __attribute__((ext_vector_type(8)))  float    v8f;

#define NUM_NODES 2048
#define BATCHSZ   2
#define NROWS     4096      // B * NUM_NODES
#define SEQ       168
#define HID       256
#define FC_IN     608       // 32 * 19
#define NHEAD     8
#define HDIM      32
#define UPART     80
#define UTOP      40
#define NBH       16        // B * NHEAD

// ---------------------------------------------------------------- zero out
__global__ void k_zero4(float4* __restrict__ out, int n4) {
    int i = blockIdx.x * 256 + threadIdx.x;
    if (i < n4) out[i] = make_float4(0.f, 0.f, 0.f, 0.f);
}

// ------------------------------------------------------- f32 -> f16 convert
__global__ void k_cvt_h(const float* __restrict__ src, _Float16* __restrict__ dst, int n) {
    int i = blockIdx.x * 256 + threadIdx.x;
    if (i < n) dst[i] = (_Float16)src[i];
}

// ------------------------------------------------- conv stack (wave/row)
// sm layout per wave (2048 floats): [0,168) input, [168,816) conv0 (8x81),
// [816,1440) conv1 (16x39), [1440,2048) conv2 (32x19)
__global__ void k_conv(const float* __restrict__ x,
                       const float* __restrict__ w0, const float* __restrict__ b0,
                       const float* __restrict__ w1, const float* __restrict__ b1,
                       const float* __restrict__ w2, const float* __restrict__ b2,
                       const float* __restrict__ g0, const float* __restrict__ be0,
                       const float* __restrict__ g1, const float* __restrict__ be1,
                       const float* __restrict__ g2, const float* __restrict__ be2,
                       _Float16* __restrict__ feat) {
    __shared__ float sm[4 * 2048];
    const int wave = threadIdx.x >> 5;
    const int lane = threadIdx.x & 31;
    const int row  = blockIdx.x * 4 + wave;
    float* s_in = &sm[wave * 2048];
    float* s0   = s_in + 168;
    float* s1   = s_in + 816;
    float* s2   = s_in + 1440;

    // stage input row via gfx1250 async global->LDS copy (ASYNCcnt tracked)
    {
        unsigned lds_base = (unsigned)(uintptr_t)(void*)s_in;
        for (int i = lane; i < SEQ; i += 32) {
            unsigned long long ga = (unsigned long long)(uintptr_t)(x + (size_t)row * SEQ + i);
            unsigned la = lds_base + (unsigned)(i * 4);
            asm volatile("global_load_async_to_lds_b32 %0, %1, off"
                         :: "v"(la), "v"(ga) : "memory");
        }
        asm volatile("s_wait_asynccnt 0" ::: "memory");
    }
    __syncthreads();

    // conv0 (1->8, k=7, s=2) + relu
    for (int o = lane; o < 8 * 81; o += 32) {
        int co = o / 81, p = o % 81;
        float acc = b0[co];
        #pragma unroll
        for (int t = 0; t < 7; ++t) acc += w0[co * 7 + t] * s_in[p * 2 + t];
        s0[o] = acc > 0.f ? acc : 0.f;
    }
    __syncthreads();
    // LN0 over 81 per channel
    if (lane < 8) {
        float m = 0.f;
        for (int j = 0; j < 81; ++j) m += s0[lane * 81 + j];
        m *= (1.f / 81.f);
        float v = 0.f;
        for (int j = 0; j < 81; ++j) { float d = s0[lane * 81 + j] - m; v += d * d; }
        float inv = rsqrtf(v * (1.f / 81.f) + 1e-5f);
        for (int j = 0; j < 81; ++j)
            s0[lane * 81 + j] = (s0[lane * 81 + j] - m) * inv * g0[j] + be0[j];
    }
    __syncthreads();

    // conv1 (8->16, k=5, s=2) + relu
    for (int o = lane; o < 16 * 39; o += 32) {
        int co = o / 39, p = o % 39;
        float acc = b1[co];
        for (int ci = 0; ci < 8; ++ci)
            #pragma unroll
            for (int t = 0; t < 5; ++t)
                acc += w1[(co * 8 + ci) * 5 + t] * s0[ci * 81 + p * 2 + t];
        s1[o] = acc > 0.f ? acc : 0.f;
    }
    __syncthreads();
    // LN1 over 39 per channel
    if (lane < 16) {
        float m = 0.f;
        for (int j = 0; j < 39; ++j) m += s1[lane * 39 + j];
        m *= (1.f / 39.f);
        float v = 0.f;
        for (int j = 0; j < 39; ++j) { float d = s1[lane * 39 + j] - m; v += d * d; }
        float inv = rsqrtf(v * (1.f / 39.f) + 1e-5f);
        for (int j = 0; j < 39; ++j)
            s1[lane * 39 + j] = (s1[lane * 39 + j] - m) * inv * g1[j] + be1[j];
    }
    __syncthreads();

    // conv2 (16->32, k=3, s=2) + relu
    for (int o = lane; o < 32 * 19; o += 32) {
        int co = o / 19, p = o % 19;
        float acc = b2[co];
        for (int ci = 0; ci < 16; ++ci)
            #pragma unroll
            for (int t = 0; t < 3; ++t)
                acc += w2[(co * 16 + ci) * 3 + t] * s1[ci * 39 + p * 2 + t];
        s2[o] = acc > 0.f ? acc : 0.f;
    }
    __syncthreads();
    // LN2 over 19 per channel (32 channels, one per lane)
    {
        int c = lane;
        float m = 0.f;
        for (int j = 0; j < 19; ++j) m += s2[c * 19 + j];
        m *= (1.f / 19.f);
        float v = 0.f;
        for (int j = 0; j < 19; ++j) { float d = s2[c * 19 + j] - m; v += d * d; }
        float inv = rsqrtf(v * (1.f / 19.f) + 1e-5f);
        for (int j = 0; j < 19; ++j)
            s2[c * 19 + j] = (s2[c * 19 + j] - m) * inv * g2[j] + be2[j];
    }
    __syncthreads();
    for (int i = lane; i < FC_IN; i += 32) feat[row * FC_IN + i] = (_Float16)s2[i];
}

// --------------------------------------- FC GEMM (4096x608 @ 608x256) WMMA
// 128-thread blocks: 4 waves, each computing one 16x16 tile (shared m0)
__global__ void k_fc(const _Float16* __restrict__ feat, const _Float16* __restrict__ fcwh,
                     const float* __restrict__ fcb, float* __restrict__ preact) {
    const int wave = threadIdx.x >> 5;
    const int lane = threadIdx.x & 31;
    const int m0 = blockIdx.x * 16;
    const int n0 = (blockIdx.y * 4 + wave) * 16;
    const int m = lane & 15;       // row index for A, col index for B tile
    const int half = lane >> 4;
    v8f acc = {};
    const _Float16* arow = feat + (size_t)(m0 + m) * FC_IN;
    const _Float16* brow = fcwh + (size_t)(n0 + m) * FC_IN;
    for (int kk = 0; kk < 19; ++kk) {
        v16h a, b;
        #pragma unroll
        for (int v = 0; v < 8; ++v) {
            int k0 = kk * 32 + (v >> 2) * 16 + half * 8 + (v & 3) * 2;
            a[2 * v]     = arow[k0];
            a[2 * v + 1] = arow[k0 + 1];
            b[2 * v]     = brow[k0];
            b[2 * v + 1] = brow[k0 + 1];
        }
        acc = __builtin_amdgcn_wmma_f32_16x16x32_f16(false, a, false, b,
                                                     (short)0, acc, false, false);
    }
    #pragma unroll
    for (int j = 0; j < 8; ++j) {
        int r = m0 + j + 8 * half;
        int c = n0 + m;
        float val = acc[j] + fcb[c];
        preact[(size_t)r * HID + c] = val > 0.f ? val : 0.f;
    }
}

// ------------------------------------------------------ LN over 256 (ln3)
__global__ void k_ln(const float* __restrict__ pre, const float* __restrict__ g,
                     const float* __restrict__ b, _Float16* __restrict__ h) {
    const int wave = threadIdx.x >> 5;
    const int lane = threadIdx.x & 31;
    const int row  = blockIdx.x * 8 + wave;
    const float* p = pre + (size_t)row * HID;
    float vals[8];
    float s = 0.f;
    #pragma unroll
    for (int j = 0; j < 8; ++j) { vals[j] = p[lane + 32 * j]; s += vals[j]; }
    for (int off = 16; off; off >>= 1) s += __shfl_xor(s, off, 32);
    float mean = s * (1.f / 256.f);
    float v = 0.f;
    #pragma unroll
    for (int j = 0; j < 8; ++j) { float d = vals[j] - mean; v += d * d; }
    for (int off = 16; off; off >>= 1) v += __shfl_xor(v, off, 32);
    float inv = rsqrtf(v * (1.f / 256.f) + 1e-5f);
    #pragma unroll
    for (int j = 0; j < 8; ++j) {
        int c = lane + 32 * j;
        h[(size_t)row * HID + c] = (_Float16)((vals[j] - mean) * inv * g[c] + b[c]);
    }
}

// ------------------------------ q/k projections (4096x256 @ 256x256) WMMA
// grid.z: 0 -> q, 1 -> k; f32 output in (b, head, l, d) + f16 copy for attn
__global__ void k_qk(const _Float16* __restrict__ h,
                     const _Float16* __restrict__ qwh, const float* __restrict__ qb,
                     const _Float16* __restrict__ kwh, const float* __restrict__ kb,
                     float* __restrict__ qo, float* __restrict__ ko,
                     _Float16* __restrict__ qoh, _Float16* __restrict__ koh) {
    const _Float16* w    = blockIdx.z ? kwh : qwh;
    const float*    bias = blockIdx.z ? kb  : qb;
    float*          out  = blockIdx.z ? ko  : qo;
    _Float16*       outh = blockIdx.z ? koh : qoh;
    const int wave = threadIdx.x >> 5;
    const int lane = threadIdx.x & 31;
    const int m0 = blockIdx.x * 16;
    const int n0 = (blockIdx.y * 4 + wave) * 16;
    const int m = lane & 15;
    const int half = lane >> 4;
    v8f acc = {};
    const _Float16* arow = h + (size_t)(m0 + m) * HID;
    const _Float16* brow = w + (size_t)(n0 + m) * HID;
    for (int kk = 0; kk < 8; ++kk) {
        v16h a, b;
        #pragma unroll
        for (int v = 0; v < 8; ++v) {
            int k0 = kk * 32 + (v >> 2) * 16 + half * 8 + (v & 3) * 2;
            a[2 * v]     = arow[k0];
            a[2 * v + 1] = arow[k0 + 1];
            b[2 * v]     = brow[k0];
            b[2 * v + 1] = brow[k0 + 1];
        }
        acc = __builtin_amdgcn_wmma_f32_16x16x32_f16(false, a, false, b,
                                                     (short)0, acc, false, false);
    }
    #pragma unroll
    for (int j = 0; j < 8; ++j) {
        int r = m0 + j + 8 * half;         // global row = b*2048 + l
        int c = n0 + m;                    // global col = head*32 + d
        int bb = r >> 11, l = r & 2047;
        int head = c >> 5, d = c & 31;
        size_t o = ((size_t)(bb * NHEAD + head) * NUM_NODES + l) * HDIM + d;
        float val = acc[j] + bias[c];
        out[o]  = val;
        outh[o] = (_Float16)val;
    }
}

// --------------------------- M = max_s(q.Ks) - sum_s(q.Ks)/L per (bh, l)
__global__ void k_msample(const float* __restrict__ q, const float* __restrict__ kmat,
                          const int* __restrict__ idx, float* __restrict__ M) {
    int t = blockIdx.x * 256 + threadIdx.x;          // 0..32767
    int bh = t >> 11, l = t & 2047;
    const float* qp = q + (size_t)(bh * NUM_NODES + l) * HDIM;
    float qv[HDIM];
    #pragma unroll
    for (int d = 0; d < HDIM; ++d) qv[d] = qp[d];
    float mx = -INFINITY, sm = 0.f;
    for (int s = 0; s < UPART; ++s) {
        int j = idx[l * UPART + s];
        const float* kp = kmat + (size_t)(bh * NUM_NODES + j) * HDIM;
        if (s + 1 < UPART)
            __builtin_prefetch(kmat + (size_t)(bh * NUM_NODES + idx[l * UPART + s + 1]) * HDIM, 0, 3);
        float dot = 0.f;
        #pragma unroll
        for (int d = 0; d < HDIM; ++d) dot += qv[d] * kp[d];
        mx = fmaxf(mx, dot);
        sm += dot;
    }
    M[t] = mx - sm * (1.f / (float)NUM_NODES);
}

// ---------------------------------------------- top-40 indices per (b,h)
__global__ void k_topk(const float* __restrict__ M, int* __restrict__ top) {
    __shared__ float vals[NUM_NODES];
    __shared__ float rv[256];
    __shared__ int   ri[256];
    int bh = blockIdx.x, tid = threadIdx.x;
    for (int i = tid; i < NUM_NODES; i += 256) vals[i] = M[bh * NUM_NODES + i];
    __syncthreads();
    for (int it = 0; it < UTOP; ++it) {
        float bm = -INFINITY; int bi = NUM_NODES;
        for (int i = tid; i < NUM_NODES; i += 256) {
            float v = vals[i];
            if (v > bm) { bm = v; bi = i; }
        }
        rv[tid] = bm; ri[tid] = bi;
        __syncthreads();
        for (int s = 128; s; s >>= 1) {
            if (tid < s) {
                if (rv[tid + s] > rv[tid] ||
                    (rv[tid + s] == rv[tid] && ri[tid + s] < ri[tid])) {
                    rv[tid] = rv[tid + s]; ri[tid] = ri[tid + s];
                }
            }
            __syncthreads();
        }
        if (tid == 0) { top[bh * UTOP + it] = ri[0]; vals[ri[0]] = -INFINITY; }
        __syncthreads();
    }
}

// -------------------- scores = (Q_reduce @ k^T) / sqrt(D), one WMMA/tile
// rows 40..47 zero-padded; grid (128 l-tiles, 3 u-tiles, 16 bh)
__global__ void k_scores(const _Float16* __restrict__ qh, const _Float16* __restrict__ kh,
                         const int* __restrict__ top, float* __restrict__ sc) {
    const int lane = threadIdx.x & 31;
    const int lt = blockIdx.x, ut = blockIdx.y, bh = blockIdx.z;
    const int m = lane & 15;
    const int half = lane >> 4;
    const int u = ut * 16 + m;
    v16h a = {}, b;
    if (u < UTOP) {
        const _Float16* qp = qh + (size_t)(bh * NUM_NODES + top[bh * UTOP + u]) * HDIM;
        #pragma unroll
        for (int v = 0; v < 8; ++v) {
            int k0 = (v >> 2) * 16 + half * 8 + (v & 3) * 2;
            a[2 * v]     = qp[k0];
            a[2 * v + 1] = qp[k0 + 1];
        }
    }
    const _Float16* kp = kh + (size_t)(bh * NUM_NODES + lt * 16 + m) * HDIM;
    #pragma unroll
    for (int v = 0; v < 8; ++v) {
        int k0 = (v >> 2) * 16 + half * 8 + (v & 3) * 2;
        b[2 * v]     = kp[k0];
        b[2 * v + 1] = kp[k0 + 1];
    }
    v8f acc = {};
    acc = __builtin_amdgcn_wmma_f32_16x16x32_f16(false, a, false, b,
                                                 (short)0, acc, false, false);
    const float scale = 0.1767766952966369f; // 1/sqrt(32)
    #pragma unroll
    for (int j = 0; j < 8; ++j) {
        int r = ut * 16 + j + 8 * half;
        int c = lt * 16 + m;
        sc[((size_t)bh * 48 + r) * NUM_NODES + c] = acc[j] * scale;
    }
}

// ----------------- softmax over 2048, threshold < 1/L, mean-over-heads add
__global__ void k_softmax_scatter(const float* __restrict__ sc, const int* __restrict__ top,
                                  float* __restrict__ out) {
    __shared__ float red[256];
    int bh = blockIdx.x, u = blockIdx.y, tid = threadIdx.x;
    const float* row = sc + ((size_t)bh * 48 + u) * NUM_NODES;
    float vals[8];
    float mx = -INFINITY;
    #pragma unroll
    for (int j = 0; j < 8; ++j) { vals[j] = row[tid + 256 * j]; mx = fmaxf(mx, vals[j]); }
    red[tid] = mx; __syncthreads();
    for (int s = 128; s; s >>= 1) {
        if (tid < s) red[tid] = fmaxf(red[tid], red[tid + s]);
        __syncthreads();
    }
    mx = red[0]; __syncthreads();
    float sm = 0.f;
    #pragma unroll
    for (int j = 0; j < 8; ++j) { vals[j] = expf(vals[j] - mx); sm += vals[j]; }
    red[tid] = sm; __syncthreads();
    for (int s = 128; s; s >>= 1) {
        if (tid < s) red[tid] += red[tid + s];
        __syncthreads();
    }
    float inv = 1.f / red[0];
    int lrow = top[bh * UTOP + u];
    int bb = bh >> 3;
    float* orow = out + ((size_t)bb * NUM_NODES + lrow) * NUM_NODES;
    const float thr = 1.f / (float)NUM_NODES;
    #pragma unroll
    for (int j = 0; j < 8; ++j) {
        float a = vals[j] * inv;
        a = (a < thr) ? 0.f : a;
        if (a != 0.f) atomicAdd(&orow[tid + 256 * j], a * 0.125f);
    }
}

// ---------------------------------------------------------------- launch
extern "C" void kernel_launch(void* const* d_in, const int* in_sizes, int n_in,
                              void* d_out, int out_size, void* d_ws, size_t ws_size,
                              hipStream_t stream) {
    const float* x   = (const float*)d_in[0];
    const float* w0  = (const float*)d_in[1];  const float* b0  = (const float*)d_in[2];
    const float* w1  = (const float*)d_in[3];  const float* b1  = (const float*)d_in[4];
    const float* w2  = (const float*)d_in[5];  const float* b2  = (const float*)d_in[6];
    const float* g0  = (const float*)d_in[7];  const float* be0 = (const float*)d_in[8];
    const float* g1  = (const float*)d_in[9];  const float* be1 = (const float*)d_in[10];
    const float* g2  = (const float*)d_in[11]; const float* be2 = (const float*)d_in[12];
    const float* g3  = (const float*)d_in[13]; const float* be3 = (const float*)d_in[14];
    const float* fcw = (const float*)d_in[15]; const float* fcb = (const float*)d_in[16];
    const float* qw  = (const float*)d_in[17]; const float* qb  = (const float*)d_in[18];
    const float* kw  = (const float*)d_in[19]; const float* kb  = (const float*)d_in[20];
    const int*   idx = (const int*)d_in[21];
    float* out = (float*)d_out;

    // workspace layout (bytes, 256-aligned); scores aliases dead feat/preact
    char* ws = (char*)d_ws;
    _Float16* feat   = (_Float16*)(ws + 0);          // 4096*608*2  = 4,980,736
    float*    preact = (float*)(ws + 4980736);       // 4096*256*4  = 4,194,304
    _Float16* hbuf   = (_Float16*)(ws + 9175040);    // 4096*256*2  = 2,097,152
    float*    qf     = (float*)(ws + 11272192);      // 16*2048*32*4= 4,194,304
    float*    kf     = (float*)(ws + 15466496);      // 4,194,304
    float*    Mf     = (float*)(ws + 19660800);      // 131,072
    int*      Mtop   = (int*)(ws + 19791872);        // 2,560
    _Float16* fcwh   = (_Float16*)(ws + 19794432);   // 256*608*2 = 311,296
    _Float16* qwh    = (_Float16*)(ws + 20105728);   // 256*256*2 = 131,072
    _Float16* kwh    = (_Float16*)(ws + 20236800);   // 131,072
    _Float16* qfh    = (_Float16*)(ws + 20367872);   // 16*2048*32*2 = 2,097,152
    _Float16* kfh    = (_Float16*)(ws + 22465024);   // 2,097,152   (end 24,562,176)
    float*    scores = (float*)(ws + 0);             // 16*48*2048*4 = 6,291,456 (alias)

    // zero output (harness poisons d_out)
    k_zero4<<<8192, 256, 0, stream>>>((float4*)out, (2 * NUM_NODES * NUM_NODES) / 4);
    // pre-convert weights to f16 (one-time; removes inner-loop cvts)
    k_cvt_h<<<(HID * FC_IN + 255) / 256, 256, 0, stream>>>(fcw, fcwh, HID * FC_IN);
    k_cvt_h<<<(HID * HID + 255) / 256, 256, 0, stream>>>(qw, qwh, HID * HID);
    k_cvt_h<<<(HID * HID + 255) / 256, 256, 0, stream>>>(kw, kwh, HID * HID);
    // conv stack (async global->LDS staging)
    k_conv<<<1024, 128, 0, stream>>>(x, w0, b0, w1, b1, w2, b2,
                                     g0, be0, g1, be1, g2, be2, feat);
    // fc GEMM (WMMA)
    k_fc<<<dim3(NROWS / 16, HID / 64), 128, 0, stream>>>(feat, fcwh, fcb, preact);
    // layer norm 3
    k_ln<<<NROWS / 8, 256, 0, stream>>>(preact, g3, be3, hbuf);
    // q / k projections (WMMA)
    k_qk<<<dim3(NROWS / 16, HID / 64, 2), 128, 0, stream>>>(hbuf, qwh, qb, kwh, kb,
                                                            qf, kf, qfh, kfh);
    // sampled M metric
    k_msample<<<(NBH * NUM_NODES) / 256, 256, 0, stream>>>(qf, kf, idx, Mf);
    // top-40 per (b,h)
    k_topk<<<NBH, 256, 0, stream>>>(Mf, Mtop);
    // sparse scores (WMMA, K = D = 32 in a single 16x16x32 step)
    k_scores<<<dim3(NUM_NODES / 16, 3, NBH), 32, 0, stream>>>(qfh, kfh, Mtop, scores);
    // softmax + threshold + head-mean scatter
    k_softmax_scatter<<<dim3(NBH, UTOP), 256, 0, stream>>>(scores, Mtop, out);
}
